// RotaryMHA_63307817943748
// MI455X (gfx1250) — matrix operand order repaired and checked by reading, not compiled, not run on hardware
//
#include <hip/hip_runtime.h>
#include <hip/hip_bf16.h>
#include <math.h>

// Problem constants (match reference)
#define D_MODEL 2048
#define N_HEAD 16
#define N_KV 4
#define HEAD_DIM 128
#define BB 2
#define TT 2048
#define KVD 512              // N_KV * HEAD_DIM
#define MROWS (BB * TT)      // 4096 flattened token rows

typedef __attribute__((ext_vector_type(16))) __bf16 v16bf;
typedef __attribute__((ext_vector_type(8)))  __bf16 v8bf;
typedef __attribute__((ext_vector_type(8)))  float  v8f;
typedef __attribute__((ext_vector_type(4)))  unsigned int u32x4;
typedef __attribute__((ext_vector_type(8)))  int i32x8;
typedef __attribute__((ext_vector_type(4)))  int i32x4;

__device__ __forceinline__ v8f zero_v8f() {
  v8f z;
#pragma unroll
  for (int i = 0; i < 8; ++i) z[i] = 0.0f;
  return z;
}

// ---------------------------------------------------------------------------
// WMMA fragment loaders (wave32); contiguous k order, the same for A and B.
//
// A fragment (16x32 bf16, M x K): lane L holds row M = L&15.
//   lanes 0-15 : K = 0..15 (elems 0..15)
//   lanes 16-31: K = 16..31
__device__ __forceinline__ v16bf load_frag_a(const __bf16* p, int ld) {
  const int lane = threadIdx.x & 31;
  const int half = lane >> 4;
  const int row  = lane & 15;
  const __bf16* q = p + (size_t)row * ld + half * 16;
  v8bf lo = *(const v8bf*)(q);
  v8bf hi = *(const v8bf*)(q + 8);
  return __builtin_shufflevector(lo, hi, 0,1,2,3,4,5,6,7,8,9,10,11,12,13,14,15);
}

// B fragment (32x16 bf16, K x N): lane L holds column N = L&15.
//   lanes 0-15 cover K = 0..15 contiguously (2 per VGPR), lanes 16-31 K = 16..31.
// Source stores column n of B contiguously along K: B[k][n] = src[n*ld + k].
__device__ __forceinline__ v16bf load_frag_b(const __bf16* p, int ld) {
  const int lane = threadIdx.x & 31;
  const int half = lane >> 4;
  const int col  = lane & 15;
  const __bf16* q = p + (size_t)col * ld + half * 16;
  v8bf lo = *(const v8bf*)(q);
  v8bf hi = *(const v8bf*)(q + 8);
  return __builtin_shufflevector(lo, hi, 0,1,2,3,4,5,6,7,8,9,10,11,12,13,14,15);
}

__device__ __forceinline__ v8f wmma_bf16(v16bf a, v16bf b, v8f c) {
  return __builtin_amdgcn_wmma_f32_16x16x32_bf16(false, a, false, b, (short)0, c,
                                                 false, false);
}

// Low 32 bits of a generic pointer into the LDS aperture == LDS byte offset
// (flat LDS addressing truncates to addr[31:0], ISA 10.2).
__device__ __forceinline__ unsigned lds_offset(const void* p) {
  return (unsigned)(size_t)p;
}

// ---------------------------------------------------------------------------
// Tensor Data Mover: DMA a 2-D bf16 tile (rows x cols, row stride in elements)
// from global memory into LDS (row-major, contiguous). Wave-level operation.
// D# layout per CDNA5 ISA.
#if defined(__has_builtin)
#if __has_builtin(__builtin_amdgcn_tensor_load_to_lds)
#define HAVE_TDM 1
#endif
#endif

#ifdef HAVE_TDM
__device__ __forceinline__ void tdm_load_tile_2d_bf16(
    unsigned ldsByteAddr, const void* gsrc, unsigned rows, unsigned cols,
    unsigned rowStrideElems) {
  const unsigned long long ga = (unsigned long long)gsrc;
  u32x4 g0;
  g0[0] = 1u;                                   // count=1, user mode
  g0[1] = ldsByteAddr;                          // lds_addr
  g0[2] = (unsigned)(ga & 0xFFFFFFFFull);       // global_addr[31:0]
  g0[3] = (unsigned)((ga >> 32) & 0x1FFFFFFull) // global_addr[56:32]
          | (2u << 30);                         // type = 2 ("image")
  i32x8 g1;
  g1[0] = (int)(1u << 16);          // wg_mask=0, data_size=1 (2 bytes)
  g1[1] = (int)(cols << 16);        // tensor_dim0[15:0] (bits 63:48)
  g1[2] = (int)(rows << 16);        // tensor_dim0 hi=0 | tensor_dim1[15:0]
  g1[3] = (int)(cols << 16);        // tensor_dim1 hi=0 | tile_dim0
  g1[4] = (int)rows;                // tile_dim1 | tile_dim2=0
  g1[5] = (int)rowStrideElems;      // tensor_dim0_stride[31:0]
  g1[6] = 0;                        // stride hi, tensor_dim1_stride lo
  g1[7] = 0;
  i32x4 zz;
  zz[0] = 0; zz[1] = 0; zz[2] = 0; zz[3] = 0;
#if __has_include(<hip/amd_detail/amd_gfx1250_TDM.h>)
  i32x8 z8;
#pragma unroll
  for (int i = 0; i < 8; ++i) z8[i] = 0;
  __builtin_amdgcn_tensor_load_to_lds(g0, g1, zz, zz, z8, 0);  // clang-23 form
#else
  __builtin_amdgcn_tensor_load_to_lds(g0, g1, zz, zz, 0);      // ROCm 7.2 form
#endif
}
#endif

// ---------------------------------------------------------------------------
// fp32 -> bf16 conversion (grid-stride, vectorized read)
__global__ void cvt_f32_to_bf16(const float* __restrict__ src,
                                __bf16* __restrict__ dst, int n) {
  const int tid    = blockIdx.x * blockDim.x + threadIdx.x;
  const int stride = gridDim.x * blockDim.x;
  for (int j = tid * 4; j < n; j += stride * 4) {
    float4 f = *(const float4*)(src + j);
    dst[j + 0] = (__bf16)f.x;
    dst[j + 1] = (__bf16)f.y;
    dst[j + 2] = (__bf16)f.z;
    dst[j + 3] = (__bf16)f.w;
  }
}

// ---------------------------------------------------------------------------
// GEMM: out[M,N] = A[M,K] @ W[N,K]^T   (bf16 in, f32 accumulate)
// Block = 128 threads = 4 waves; each wave computes a 32x64 tile (2x4 frags).
// Block tile = 64 rows x 128 cols. Grid = (N/128, M/64).
template <int N, int K, bool F32OUT>
__global__ __launch_bounds__(128)
void gemm_xwt(const __bf16* __restrict__ A, const __bf16* __restrict__ W,
              void* __restrict__ outp) {
  const int wave = threadIdx.x >> 5;
  const int lane = threadIdx.x & 31;
  const int half = lane >> 4;
  const int cidx = lane & 15;
  const int mBase = blockIdx.y * 64  + (wave & 1) * 32;
  const int nBase = blockIdx.x * 128 + (wave >> 1) * 64;

  v8f acc[2][4];
#pragma unroll
  for (int mt = 0; mt < 2; ++mt)
#pragma unroll
    for (int nt = 0; nt < 4; ++nt) acc[mt][nt] = zero_v8f();

  for (int k0 = 0; k0 < K; k0 += 32) {
    // Prefetch ~8 K-steps ahead (global_prefetch_b8).
    if (k0 + 256 < K) {
      __builtin_prefetch(A + (size_t)mBase * K + k0 + 256, 0, 1);
      __builtin_prefetch(W + (size_t)nBase * K + k0 + 256, 0, 1);
    }
    v16bf a0 = load_frag_a(A + (size_t)mBase * K + k0, K);
    v16bf a1 = load_frag_a(A + (size_t)(mBase + 16) * K + k0, K);
    v16bf b[4];
#pragma unroll
    for (int nt = 0; nt < 4; ++nt)
      b[nt] = load_frag_b(W + (size_t)(nBase + nt * 16) * K + k0, K);
#pragma unroll
    for (int nt = 0; nt < 4; ++nt) {
      acc[0][nt] = wmma_bf16(a0, b[nt], acc[0][nt]);
      acc[1][nt] = wmma_bf16(a1, b[nt], acc[1][nt]);
    }
  }

  // C/D layout: row m = tile + half*8 + r, col n = tile + (lane&15)
#pragma unroll
  for (int mt = 0; mt < 2; ++mt) {
#pragma unroll
    for (int nt = 0; nt < 4; ++nt) {
      const size_t base =
          (size_t)(mBase + mt * 16 + half * 8) * N + nBase + nt * 16 + cidx;
      if constexpr (F32OUT) {
        float* out = (float*)outp;
#pragma unroll
        for (int r = 0; r < 8; ++r) out[base + (size_t)r * N] = acc[mt][nt][r];
      } else {
        __bf16* out = (__bf16*)outp;
#pragma unroll
        for (int r = 0; r < 8; ++r)
          out[base + (size_t)r * N] = (__bf16)acc[mt][nt][r];
      }
    }
  }
}

// ---------------------------------------------------------------------------
// Causal GQA flash attention.
// Grid = (T/64, N_HEAD, B); block = 128 threads = 4 waves.
// Wave w owns 16 query rows [qBlock + 16w, +16). Key tiles of 32, online
// softmax. Staging per tile:
//   K tile : TDM tensor_load_to_lds by wave 0 (TENSORcnt fenced), or async
//            global->LDS fallback.
//   V tile : async global->LDS (ASYNCcnt fenced) row-major, then LDS->LDS
//            transpose so PV B-fragments are contiguous ds_load_b128.
__global__ __launch_bounds__(128)
void attn_fwd(const __bf16* __restrict__ Q, const __bf16* __restrict__ Kc,
              const __bf16* __restrict__ Vc, __bf16* __restrict__ AO) {
  const int qBlock = blockIdx.x * 64;
  const int h      = blockIdx.y;
  const int bat    = blockIdx.z;
  const int kvh    = h >> 2;                 // GQA: 4 heads per KV head
  const int wave   = threadIdx.x >> 5;
  const int lane   = threadIdx.x & 31;
  const int half   = lane >> 4;
  const int l16    = lane & 15;
  const int qStart = qBlock + wave * 16;
  const float scale = 0.08838834764831845f;  // 1/sqrt(128)

  __shared__ __align__(16) __bf16 Krm[32 * HEAD_DIM];  // K tile row-major [k][d]
  __shared__ __align__(16) __bf16 Vrm[32 * HEAD_DIM];  // V tile row-major [k][d]
  __shared__ __align__(16) __bf16 Vt[HEAD_DIM * 32];   // V tile transposed [d][k]
  __shared__ __align__(16) __bf16 Pb[4][16 * 32];      // per-wave P scratch

  // Resident Q fragments: 16 rows x 128 d = 4 fragments of 16x32
  const __bf16* qbase =
      Q + ((size_t)(bat * TT + qStart)) * D_MODEL + h * HEAD_DIM;
  v16bf qf[4];
#pragma unroll
  for (int kc = 0; kc < 4; ++kc) qf[kc] = load_frag_a(qbase + kc * 32, D_MODEL);

  v8f o[8];
#pragma unroll
  for (int dt = 0; dt < 8; ++dt) o[dt] = zero_v8f();
  float mrun[8], lrun[8];
#pragma unroll
  for (int r = 0; r < 8; ++r) { mrun[r] = -INFINITY; lrun[r] = 0.0f; }

  // Per-thread staging assignment: 4 threads per key row, 32 d-values each.
  const int sKey = threadIdx.x >> 2;
  const int sD0  = (threadIdx.x & 3) * 32;
  const unsigned vrmOff = lds_offset(&Vrm[sKey * HEAD_DIM + sD0]);
#ifndef HAVE_TDM
  const unsigned krmOff = lds_offset(&Krm[sKey * HEAD_DIM + sD0]);
#endif

  const int nTiles = qBlock / 32 + 2;  // uniform across the block (causal bound)
  for (int t = 0; t < nTiles; ++t) {
    const int kBase = t * 32;
    __syncthreads();  // previous iteration's LDS consumers are done

    // ---- K tile (32 keys x 128 d) global -> LDS, row-major ----
#ifdef HAVE_TDM
    if (wave == 0) {
      const __bf16* ktile =
          Kc + ((size_t)(bat * TT + kBase)) * KVD + kvh * HEAD_DIM;
      tdm_load_tile_2d_bf16(lds_offset(&Krm[0]), ktile, 32, HEAD_DIM, KVD);
    }
#else
    {
      const __bf16* ksrc =
          Kc + ((size_t)(bat * TT + kBase + sKey)) * KVD + kvh * HEAD_DIM + sD0;
#pragma unroll
      for (int j = 0; j < 4; ++j) {
        asm volatile("global_load_async_to_lds_b128 %0, %1, off"
                     :
                     : "v"(krmOff + j * 16), "v"(ksrc + j * 8)
                     : "memory");
      }
    }
#endif

    // ---- V tile global -> LDS (async DMA), row-major staging ----
    {
      const __bf16* vsrc =
          Vc + ((size_t)(bat * TT + kBase + sKey)) * KVD + kvh * HEAD_DIM + sD0;
#pragma unroll
      for (int j = 0; j < 4; ++j) {
        asm volatile("global_load_async_to_lds_b128 %0, %1, off"
                     :
                     : "v"(vrmOff + j * 16), "v"(vsrc + j * 8)
                     : "memory");
      }
    }

    asm volatile("s_wait_asynccnt 0x0" ::: "memory");  // V (and async K) landed
#ifdef HAVE_TDM
    if (wave == 0) __builtin_amdgcn_s_wait_tensorcnt(0);  // K tile landed
#endif
    __syncthreads();  // Krm + Vrm visible to all waves

    // ---- LDS->LDS transpose of V tile: Vrm[k][d] -> Vt[d][k] ----
    {
#pragma unroll
      for (int j = 0; j < 4; ++j) {
        v8bf v = *(const v8bf*)(&Vrm[sKey * HEAD_DIM + sD0 + j * 8]);
#pragma unroll
        for (int e = 0; e < 8; ++e) Vt[(sD0 + j * 8 + e) * 32 + sKey] = v[e];
      }
    }
    __syncthreads();  // Vt visible

    // ---- S = Q @ K^T from LDS: two 16x16 f32 fragments ----
    v8f s0 = zero_v8f(), s1 = zero_v8f();
#pragma unroll
    for (int kc = 0; kc < 4; ++kc) {
      v16bf kb0 = load_frag_b(&Krm[kc * 32], HEAD_DIM);            // keys 0..15
      v16bf kb1 = load_frag_b(&Krm[16 * HEAD_DIM + kc * 32], HEAD_DIM); // 16..31
      s0 = wmma_bf16(qf[kc], kb0, s0);
      s1 = wmma_bf16(qf[kc], kb1, s1);
    }

    // ---- Online softmax (rows live in 16-lane halves; xor 1/2/4/8) ----
    float corr[8];
    const int key0 = kBase + l16;
    const int key1 = key0 + 16;
#pragma unroll
    for (int r = 0; r < 8; ++r) {
      const int qi = qStart + half * 8 + r;
      float v0 = (key0 <= qi) ? s0[r] * scale : -INFINITY;
      float v1 = (key1 <= qi) ? s1[r] * scale : -INFINITY;
      float mx = fmaxf(v0, v1);
#pragma unroll
      for (int off = 1; off < 16; off <<= 1)
        mx = fmaxf(mx, __shfl_xor(mx, off, 32));
      const float mN = fmaxf(mrun[r], mx);
      const float c  = __expf(mrun[r] - mN);
      const float p0 = __expf(v0 - mN);
      const float p1 = __expf(v1 - mN);
      float rs = p0 + p1;
#pragma unroll
      for (int off = 1; off < 16; off <<= 1) rs += __shfl_xor(rs, off, 32);
      lrun[r] = lrun[r] * c + rs;
      mrun[r] = mN;
      corr[r] = c;
      // Spill P (C-layout) to this wave's LDS scratch; LDS ops from one wave
      // complete in order, so the A-layout re-read below needs no barrier.
      Pb[wave][(half * 8 + r) * 32 + l16]      = (__bf16)p0;
      Pb[wave][(half * 8 + r) * 32 + 16 + l16] = (__bf16)p1;
    }
#pragma unroll
    for (int dt = 0; dt < 8; ++dt)
#pragma unroll
      for (int r = 0; r < 8; ++r) o[dt][r] *= corr[r];

    // ---- O += P @ V ----
    v16bf pf = load_frag_a(&Pb[wave][0], 32);
#pragma unroll
    for (int dt = 0; dt < 8; ++dt) {
      v16bf vf = load_frag_b(&Vt[(dt * 16) * 32], 32);
      o[dt] = wmma_bf16(pf, vf, o[dt]);
    }
  }

  // Normalize and store attention output (bf16) for the Wo GEMM.
  float inv[8];
#pragma unroll
  for (int r = 0; r < 8; ++r) inv[r] = 1.0f / lrun[r];
  __bf16* ao =
      AO + ((size_t)(bat * TT + qStart + half * 8)) * D_MODEL + h * HEAD_DIM;
#pragma unroll
  for (int dt = 0; dt < 8; ++dt)
#pragma unroll
    for (int r = 0; r < 8; ++r)
      ao[(size_t)r * D_MODEL + dt * 16 + l16] = (__bf16)(o[dt][r] * inv[r]);
}

// ---------------------------------------------------------------------------
extern "C" void kernel_launch(void* const* d_in, const int* in_sizes, int n_in,
                              void* d_out, int out_size, void* d_ws,
                              size_t ws_size, hipStream_t stream) {
  const float* x  = (const float*)d_in[0];
  const float* Wq = (const float*)d_in[1];
  const float* Wk = (const float*)d_in[2];
  const float* Wv = (const float*)d_in[3];
  const float* Wo = (const float*)d_in[4];
  float* out = (float*)d_out;

  // Workspace carve-up (all bf16, all 16B-aligned): ~76 MB total.
  __bf16* p  = (__bf16*)d_ws;
  __bf16* xb = p;  p += (size_t)MROWS * D_MODEL;    // x              (8.4M)
  __bf16* wq = p;  p += (size_t)D_MODEL * D_MODEL;  // Wq             (4.2M)
  __bf16* wk = p;  p += (size_t)KVD * D_MODEL;      // Wk             (1.0M)
  __bf16* wv = p;  p += (size_t)KVD * D_MODEL;      // Wv             (1.0M)
  __bf16* wo = p;  p += (size_t)D_MODEL * D_MODEL;  // Wo             (4.2M)
  __bf16* Qb = p;  p += (size_t)MROWS * D_MODEL;    // Q              (8.4M)
  __bf16* Kb = p;  p += (size_t)MROWS * KVD;        // K              (2.1M)
  __bf16* Vb = p;  p += (size_t)MROWS * KVD;        // V              (2.1M)
  __bf16* AOb = p;                                  // attn out       (8.4M)

  const dim3 cvtGrid(2048), cvtBlk(256);
  cvt_f32_to_bf16<<<cvtGrid, cvtBlk, 0, stream>>>(x,  xb, MROWS * D_MODEL);
  cvt_f32_to_bf16<<<cvtGrid, cvtBlk, 0, stream>>>(Wq, wq, D_MODEL * D_MODEL);
  cvt_f32_to_bf16<<<cvtGrid, cvtBlk, 0, stream>>>(Wk, wk, KVD * D_MODEL);
  cvt_f32_to_bf16<<<cvtGrid, cvtBlk, 0, stream>>>(Wv, wv, KVD * D_MODEL);
  cvt_f32_to_bf16<<<cvtGrid, cvtBlk, 0, stream>>>(Wo, wo, D_MODEL * D_MODEL);

  const dim3 blk(128);
  // Projections: Q = x@Wq^T, K = x@Wk^T, V = x@Wv^T (bf16 out for attention)
  gemm_xwt<D_MODEL, D_MODEL, false>
      <<<dim3(D_MODEL / 128, MROWS / 64), blk, 0, stream>>>(xb, wq, Qb);
  gemm_xwt<KVD, D_MODEL, false>
      <<<dim3(KVD / 128, MROWS / 64), blk, 0, stream>>>(xb, wk, Kb);
  gemm_xwt<KVD, D_MODEL, false>
      <<<dim3(KVD / 128, MROWS / 64), blk, 0, stream>>>(xb, wv, Vb);

  // Causal GQA flash attention
  attn_fwd<<<dim3(TT / 64, N_HEAD, BB), blk, 0, stream>>>(Qb, Kb, Vb, AOb);

  // Output projection: y = attn_out @ Wo^T (fp32 out)
  gemm_xwt<D_MODEL, D_MODEL, true>
      <<<dim3(D_MODEL / 128, MROWS / 64), blk, 0, stream>>>(AOb, wo, out);
}
